// RWKV5TimeMix_71622874628583
// MI455X (gfx1250) — compile-verified, matrix operand order checked by reading, not run on hardware
//
#include <hip/hip_runtime.h>

// RWKV5 TimeMix for MI455X (gfx1250, wave32).
// Pipeline: [f32->bf16 weights] -> 4x (lerp->bf16, WMMA GEMM) -> WKV scan ->
//           groupnorm+silu gate (bf16) -> WMMA GEMM (W_o) -> d_out.
// GEMMs: v_wmma_f32_16x16x32_bf16, 64x64 wave tile (16 WMMA / K-step / wave),
// double-buffered LDS fed by GLOBAL_LOAD_ASYNC_TO_LDS_B128 (ASYNCcnt tracked).

#define B_ 2
#define T_ 4096
#define C_ 2048
#define H_ 32
#define K_ 64
#define M_TOT (B_ * T_)        // 8192 rows
#define NELT (M_TOT * C_)      // 16,777,216

typedef __attribute__((ext_vector_type(16))) __bf16 bf16x16;
typedef __attribute__((ext_vector_type(8)))  float  f32x8;
typedef int v4i __attribute__((vector_size(16)));
typedef __attribute__((address_space(1))) v4i* gas_v4i;   // global (AS1)
typedef __attribute__((address_space(3))) v4i* las_v4i;   // LDS (AS3)

#if defined(__has_builtin)
#if __has_builtin(__builtin_amdgcn_global_load_async_to_lds_b128)
#define USE_ASYNC_COPY 1
#endif
#endif
#ifndef USE_ASYNC_COPY
#define USE_ASYNC_COPY 0
#endif

__device__ __forceinline__ unsigned short f2bf(float f) {
  unsigned int u = __float_as_uint(f);
  u += 0x7fffu + ((u >> 16) & 1u);       // round-to-nearest-even
  return (unsigned short)(u >> 16);
}

// ---------------- f32 -> bf16 (weights) ----------------
__global__ __launch_bounds__(256) void k_f32_to_bf16(const float* __restrict__ in,
                                                     unsigned short* __restrict__ out,
                                                     int n) {
  int i = blockIdx.x * 256 + threadIdx.x;
  if (i < n) out[i] = f2bf(in[i]);
}

// ---------------- token-shift lerp -> bf16 ----------------
__global__ __launch_bounds__(256) void k_lerp_bf16(const float* __restrict__ x,
                                                   const float* __restrict__ shift_in,
                                                   const float* __restrict__ mix,
                                                   unsigned short* __restrict__ out) {
  int i = blockIdx.x * 256 + threadIdx.x;       // over B*T*C
  int c = i & (C_ - 1);
  int t = (i >> 11) & (T_ - 1);                 // C = 2^11
  int b = i >> 23;                              // T*C = 2^23
  float xv = x[i];
  float xx = (t == 0) ? shift_in[b * C_ + c] : x[i - C_];
  float m = mix[c];
  out[i] = f2bf(xv * m + xx * (1.0f - m));
}

// ---------------- WMMA bf16 GEMM:  C[M,N] = A[M,K] * Bw[N,K]^T ----------------
// 128 threads (4 waves, 2x2 wave grid), block tile 128x128, wave tile 64x64.
// Double-buffered LDS (48KB) with async global->LDS copies.
#define GEMM_THREADS 128
#define TM 128
#define TN 128
#define TK 32
#define LSTR 48   // halfs per LDS row (32 data + 16 pad) -> 96B, keeps 32B alignment

__global__ __launch_bounds__(GEMM_THREADS) void k_gemm_bf16(
    const unsigned short* __restrict__ A,   // [M,K] bf16 row-major
    const unsigned short* __restrict__ Bw,  // [N,K] bf16 row-major (B^T form)
    float* __restrict__ Cc,                 // [M,N] f32
    int M, int N, int Kd) {
  __shared__ __align__(32) unsigned short As[2][TM * LSTR];
  __shared__ __align__(32) unsigned short Bs[2][TN * LSTR];

  const int tid  = threadIdx.x;
  const int lane = tid & 31;
  const int wave = tid >> 5;          // 0..3
  const int wM = (wave >> 1) * 64;    // 0 or 64
  const int wN = (wave & 1) * 64;     // 0 or 64
  const int bM = blockIdx.y * TM;
  const int bN = blockIdx.x * TN;

  const int hl   = lane & 15;         // row-in-tile / col-in-tile
  const int kSel = (lane & 16);       // lanes 0-15 take K 0..15, 16-31 take 16..31
                                      // (same K permutation for A and B => correct)

  // global->LDS staging: 16B chunks; each tile = 128 rows x 64B = 512 chunks,
  // 4 chunks per thread. chunk c -> row c>>2, half-offset (c&3)*8.
  int rw[4], of[4];
#pragma unroll
  for (int j = 0; j < 4; ++j) {
    int c = j * GEMM_THREADS + tid;
    rw[j] = c >> 2;
    of[j] = (c & 3) << 3;
  }

  f32x8 acc[4][4];
#pragma unroll
  for (int mt = 0; mt < 4; ++mt)
#pragma unroll
    for (int nt = 0; nt < 4; ++nt)
      acc[mt][nt] = {};

#if !USE_ASYNC_COPY
  uint4 rA[4], rB[4];
#endif

  auto stage = [&](int buf, int k0) {
#if USE_ASYNC_COPY
#pragma unroll
    for (int j = 0; j < 4; ++j) {
      const unsigned short* gA = A  + (size_t)(bM + rw[j]) * Kd + k0 + of[j];
      const unsigned short* gB = Bw + (size_t)(bN + rw[j]) * Kd + k0 + of[j];
      __builtin_amdgcn_global_load_async_to_lds_b128(
          (gas_v4i)(void*)gA,
          (las_v4i)(void*)&As[buf][rw[j] * LSTR + of[j]], 0, 0);
      __builtin_amdgcn_global_load_async_to_lds_b128(
          (gas_v4i)(void*)gB,
          (las_v4i)(void*)&Bs[buf][rw[j] * LSTR + of[j]], 0, 0);
    }
#endif
  };
  auto load_regs = [&](int k0) {
#if !USE_ASYNC_COPY
#pragma unroll
    for (int j = 0; j < 4; ++j) {
      rA[j] = *(const uint4*)(A  + (size_t)(bM + rw[j]) * Kd + k0 + of[j]);
      rB[j] = *(const uint4*)(Bw + (size_t)(bN + rw[j]) * Kd + k0 + of[j]);
    }
#endif
  };
  auto store_lds = [&](int buf) {
#if !USE_ASYNC_COPY
#pragma unroll
    for (int j = 0; j < 4; ++j) {
      *(uint4*)&As[buf][rw[j] * LSTR + of[j]] = rA[j];
      *(uint4*)&Bs[buf][rw[j] * LSTR + of[j]] = rB[j];
    }
#endif
  };

  const int nIt = Kd / TK;

  // prologue: stage tile 0 into buffer 0
#if USE_ASYNC_COPY
  stage(0, 0);
  asm volatile("s_wait_asynccnt 0x0" ::: "memory");
  __syncthreads();
#else
  load_regs(0);
  store_lds(0);
  __syncthreads();
#endif

  for (int it = 0; it < nIt; ++it) {
    const int cur = it & 1;
    const bool more = (it + 1 < nIt);

#if USE_ASYNC_COPY
    if (more) stage(cur ^ 1, (it + 1) * TK);   // DMA overlaps the WMMAs below
#else
    if (more) load_regs((it + 1) * TK);        // loads in flight during WMMAs
#endif

    bf16x16 bfr[4];
#pragma unroll
    for (int nt = 0; nt < 4; ++nt)
      bfr[nt] = *(const bf16x16*)&Bs[cur][(wN + nt * 16 + hl) * LSTR + kSel];
#pragma unroll
    for (int mt = 0; mt < 4; ++mt) {
      bf16x16 afr = *(const bf16x16*)&As[cur][(wM + mt * 16 + hl) * LSTR + kSel];
#pragma unroll
      for (int nt = 0; nt < 4; ++nt)
        acc[mt][nt] = __builtin_amdgcn_wmma_f32_16x16x32_bf16(
            false, afr, false, bfr[nt], (short)0, acc[mt][nt], false, false);
    }

#if USE_ASYNC_COPY
    asm volatile("s_wait_asynccnt 0x0" ::: "memory");
    __syncthreads();
#else
    if (more) store_lds(cur ^ 1);  // prev-iter barrier guarantees readers done
    __syncthreads();
#endif
  }

  // C/D layout: VGPR v -> M=v (lanes 0-15) or M=v+8 (lanes 16-31); N = lane&15.
  const int rSel = (lane & 16) >> 1;       // 0 or 8
#pragma unroll
  for (int mt = 0; mt < 4; ++mt)
#pragma unroll
    for (int nt = 0; nt < 4; ++nt) {
      int col = bN + wN + nt * 16 + hl;
#pragma unroll
      for (int vv = 0; vv < 8; ++vv) {
        int row = bM + wM + mt * 16 + rSel + vv;
        Cc[(size_t)row * N + col] = acc[mt][nt][vv];
      }
    }
}

// ---------------- sequential WKV scan ----------------
// One 64-thread block per (b,h). Thread owns one V column: S[k][v_me] in 64 regs.
// y_v = sum_k r_k*S[k][v] + v_v * sum_k r_k*u_k*k_k ;  S[k][v] = w_k*S[k][v] + k_k*v_v
// NOTE: y aliases r (read-before-write per index per thread) -> no __restrict on r/y.
__global__ __launch_bounds__(64) void k_wkv_scan(
    const float* r, const float* __restrict__ k, const float* __restrict__ v,
    const float* __restrict__ decay, const float* __restrict__ faaaa,
    const float* __restrict__ s_in, float* y, float* __restrict__ s_out) {
  const int bh = blockIdx.x;             // b*H + h
  const int b  = bh >> 5;
  const int h  = bh & (H_ - 1);
  const int tid = threadIdx.x;           // v index 0..63
  __shared__ float4 q[K_];               // per-step broadcast {r, k, r*u*k, w}

  float S[K_];
  const size_t sbase = (size_t)bh * K_ * K_;
#pragma unroll
  for (int kk = 0; kk < K_; ++kk) S[kk] = s_in[sbase + kk * K_ + tid];

  const float u_me = faaaa[h * K_ + tid];
  const float w_me = __expf(-__expf(decay[h * K_ + tid]));

  size_t base = (size_t)b * T_ * C_ + h * K_;
  for (int t = 0; t < T_; ++t, base += C_) {
    float rv = r[base + tid], kv = k[base + tid], vv = v[base + tid];
    __syncthreads();
    q[tid] = make_float4(rv, kv, rv * u_me * kv, w_me);
    __syncthreads();
    float y1 = 0.0f, y2 = 0.0f;
#pragma unroll
    for (int kk = 0; kk < K_; ++kk) {
      float4 qq = q[kk];                 // single ds_load_b128 broadcast
      y1 = fmaf(qq.x, S[kk], y1);        // uses PRE-update state
      y2 += qq.z;
      S[kk] = fmaf(qq.w, S[kk], qq.y * vv);
    }
    y[base + tid] = fmaf(y2, vv, y1);
  }
#pragma unroll
  for (int kk = 0; kk < K_; ++kk) s_out[sbase + kk * K_ + tid] = S[kk];
}

// ---------------- per-head groupnorm + silu gate -> bf16 ----------------
// One wave per (b,t,h); 2 elements per lane; wave32 shuffle reduction.
__global__ __launch_bounds__(256) void k_gnorm_gate(
    const float* __restrict__ y, const float* __restrict__ g,
    const float* __restrict__ gamma, const float* __restrict__ beta,
    unsigned short* __restrict__ out) {
  const int lane = threadIdx.x & 31;
  const int waveId = blockIdx.x * 8 + (threadIdx.x >> 5);  // 0 .. B*T*H-1
  const int h = waveId & (H_ - 1);
  const size_t bt = (size_t)(waveId >> 5);                 // 0 .. B*T-1
  const size_t base = bt * C_ + h * K_;
  const int i0 = lane * 2;

  float2 yv = *(const float2*)&y[base + i0];
  float a = yv.x * 0.125f, c = yv.y * 0.125f;              // / HEAD_SIZE_DIVISOR
  float s = a + c, ss = a * a + c * c;
#pragma unroll
  for (int off = 16; off >= 1; off >>= 1) {
    s  += __shfl_xor(s, off, 32);
    ss += __shfl_xor(ss, off, 32);
  }
  float mu  = s * (1.0f / 64.0f);
  float var = ss * (1.0f / 64.0f) - mu * mu;
  float inv = rsqrtf(var + 1e-5f);

  int gi = h * K_ + i0;
  float g0 = g[base + i0], g1 = g[base + i0 + 1];
  float sg0 = g0 / (1.0f + __expf(-g0));
  float sg1 = g1 / (1.0f + __expf(-g1));
  out[base + i0]     = f2bf(((a - mu) * inv * gamma[gi]     + beta[gi])     * sg0);
  out[base + i0 + 1] = f2bf(((c - mu) * inv * gamma[gi + 1] + beta[gi + 1]) * sg1);
}

// ---------------- shift_state_out = x[:, -1] ----------------
__global__ __launch_bounds__(256) void k_shift_out(const float* __restrict__ x,
                                                   float* __restrict__ out) {
  int i = blockIdx.x * 256 + threadIdx.x;       // B*C
  int b = i >> 11, c = i & (C_ - 1);
  out[i] = x[(size_t)(b * T_ + (T_ - 1)) * C_ + c];
}

extern "C" void kernel_launch(void* const* d_in, const int* in_sizes, int n_in,
                              void* d_out, int out_size, void* d_ws, size_t ws_size,
                              hipStream_t stream) {
  const float* x        = (const float*)d_in[0];
  const float* shift_in = (const float*)d_in[1];
  const float* wkv_in   = (const float*)d_in[2];
  const float* mix_k    = (const float*)d_in[3];
  const float* mix_v    = (const float*)d_in[4];
  const float* mix_r    = (const float*)d_in[5];
  const float* mix_g    = (const float*)d_in[6];
  const float* decay    = (const float*)d_in[7];
  const float* faaaa    = (const float*)d_in[8];
  const float* Wrkvg[4] = {(const float*)d_in[9], (const float*)d_in[10],
                           (const float*)d_in[11], (const float*)d_in[12]};
  const float* W_o      = (const float*)d_in[13];
  const float* gamma    = (const float*)d_in[14];
  const float* beta     = (const float*)d_in[15];

  float* out_main  = (float*)d_out;                 // [B,T,C]
  float* out_shift = out_main + (size_t)NELT;       // [B,C]
  float* out_state = out_shift + (size_t)B_ * C_;   // [B,H,K,K]

  // workspace layout (~328 MB), with aliasing: y reuses r_f, gated reuses act_bf
  const size_t WN = (size_t)C_ * C_;                // 4,194,304 elts per weight
  unsigned short* wbf    = (unsigned short*)d_ws;   // 4 projections, bf16
  unsigned short* wo_bf  = wbf + 4 * WN;
  unsigned short* act_bf = wo_bf + WN;              // [8192,2048] bf16 (reused as gated)
  float* r_f = (float*)(act_bf + (size_t)NELT);     // also y
  float* k_f = r_f + (size_t)NELT;
  float* v_f = k_f + (size_t)NELT;
  float* g_f = v_f + (size_t)NELT;

  const dim3 gemmGrid(C_ / TN, M_TOT / TM);         // 16 x 64
  const int convBlocks = (int)(WN / 256);           // 16384
  const int eltBlocks  = NELT / 256;                // 65536

  // 1) weights -> bf16
  for (int p = 0; p < 4; ++p)
    k_f32_to_bf16<<<convBlocks, 256, 0, stream>>>(Wrkvg[p], wbf + (size_t)p * WN, (int)WN);
  k_f32_to_bf16<<<convBlocks, 256, 0, stream>>>(W_o, wo_bf, (int)WN);

  // 2) four lerp + GEMM passes (act buffer reused serially)
  const float* mixes[4] = {mix_r, mix_k, mix_v, mix_g};
  float* projs[4] = {r_f, k_f, v_f, g_f};
  for (int p = 0; p < 4; ++p) {
    k_lerp_bf16<<<eltBlocks, 256, 0, stream>>>(x, shift_in, mixes[p], act_bf);
    k_gemm_bf16<<<gemmGrid, GEMM_THREADS, 0, stream>>>(act_bf, wbf + (size_t)p * WN,
                                                       projs[p], M_TOT, C_, C_);
  }

  // 3) sequential WKV scan (y written in-place over r_f; safe per-index RAW order)
  k_wkv_scan<<<B_ * H_, K_, 0, stream>>>(r_f, k_f, v_f, decay, faaaa, wkv_in,
                                         r_f, out_state);

  // 4) groupnorm + silu gate -> bf16 (into act_bf)
  k_gnorm_gate<<<(B_ * T_ * H_) / 8, 256, 0, stream>>>(r_f, g_f, gamma, beta, act_bf);

  // 5) output projection
  k_gemm_bf16<<<gemmGrid, GEMM_THREADS, 0, stream>>>(act_bf, wo_bf, out_main,
                                                     M_TOT, C_, C_);

  // 6) shift state out
  k_shift_out<<<(B_ * C_) / 256, 256, 0, stream>>>(x, out_shift);
}
